// FourierSynthesisInput_39436389712350
// MI455X (gfx1250) — matrix operand order repaired and checked by reading, not compiled, not run on hardware
//
#include <hip/hip_runtime.h>
#include <hip/hip_bf16.h>

typedef __attribute__((ext_vector_type(16))) _Float16 v16h;
typedef __attribute__((ext_vector_type(8)))  _Float16 v8h;
typedef __attribute__((ext_vector_type(2)))  _Float16 v2h;
typedef __attribute__((ext_vector_type(8)))  float    v8f;
typedef __attribute__((ext_vector_type(4)))  float    v4f;

#define W_DIM   512
#define CH      512
#define SIZE    64
#define NBATCH  32
#define TWO_PI  6.28318530717958647692f
#define GAIN    0.044194173824159216f   // 1/sqrt(512), also 1/sqrt(CH)
#define LDS_STRIDE 520                   // 512 + 8 f16 pad -> conflict-free b128 reads

// ---------------- Phase 1: per-(n,c) params: {frx, fry, phase, amp} ----------------
__global__ __launch_bounds__(256) void prep_params(
    const float* __restrict__ w, const float* __restrict__ aw,
    const float* __restrict__ ab, const float* __restrict__ freqs,
    const float* __restrict__ phases, float4* __restrict__ params)
{
    __shared__ float red[4][64];
    __shared__ float t4[4];
    const int nb  = blockIdx.x;
    const int tid = threadIdx.x;
    {   // t = w @ (affine_w.T * gain) + affine_b   (4 dot products of length 512)
        int j = tid >> 6, l = tid & 63;
        float s = 0.f;
        for (int d = l; d < W_DIM; d += 64) s += w[nb*W_DIM + d] * aw[j*W_DIM + d];
        red[j][l] = s;
    }
    __syncthreads();
    if (tid < 4) {
        float s = 0.f;
        for (int l = 0; l < 64; ++l) s += red[tid][l];
        t4[tid] = s * GAIN + ab[tid];
    }
    __syncthreads();
    const float t0 = t4[0], t1 = t4[1], t2 = t4[2], t3 = t4[3];
    const float inv = rsqrtf(t0*t0 + t1*t1);
    const float cc = t0*inv, ss = t1*inv, tx = t2*inv, ty = t3*inv;
    const float trX = -cc*tx + ss*ty;
    const float trY = -ss*tx - cc*ty;
    for (int c = tid; c < CH; c += 256) {
        float fx  = freqs[2*c], fy = freqs[2*c+1];
        float frx =  fx*cc + fy*ss;
        float fry = -fx*ss + fy*cc;
        float ph  = phases[c] + fx*trX + fy*trY;
        float rr  = sqrtf(frx*frx + fry*fry);
        float amp = 1.0f - (rr - 2.0f) * (1.0f/30.0f);   // (|fr|-BW)/(SR/2-BW)
        amp = fminf(fmaxf(amp, 0.f), 1.f);
        params[nb*CH + c] = make_float4(frx, fry, ph, amp);
    }
}

// ---------------- Phase 2: weight -> f16 * (1/sqrt(CH)) ----------------
__global__ __launch_bounds__(256) void conv_weight(
    const float* __restrict__ wt, _Float16* __restrict__ wh)
{
    int i = blockIdx.x * 256 + threadIdx.x;
    wh[i] = (_Float16)(wt[i] * GAIN);
}

// ---------------- Phase 3: fused sin-feature generation + WMMA GEMM ----------------
// grid: NBATCH*128 blocks; block = 256 threads (8 wave32).
// Each block: batch nb, 32 consecutive pixels. Wave wv owns N-range [wv*64, wv*64+64).
__global__ __launch_bounds__(256) void synth_gemm(
    const float4* __restrict__ params, const _Float16* __restrict__ wh,
    float* __restrict__ out)
{
    __shared__ _Float16 feat[32 * LDS_STRIDE];
    const int tid  = threadIdx.x;
    const int nb   = blockIdx.x >> 7;
    const int mblk = blockIdx.x & 127;

    // ---- feature tile: 32 pixels x 512 channels, f16 into LDS ----
    const float4 P0 = params[nb*CH + 2*tid];
    const float4 P1 = params[nb*CH + 2*tid + 1];
    #pragma unroll 4
    for (int p = 0; p < 32; ++p) {
        int pg = mblk*32 + p;
        int hh = pg >> 6, ww = pg & 63;
        float gx = ((float)(2*ww + 1) * (1.0f/64.0f) - 1.0f) * 0.5f;  // xs[w]
        float gy = ((float)(2*hh + 1) * (1.0f/64.0f) - 1.0f) * 0.5f;  // ys[h]
        float v0 = __sinf((gx*P0.x + gy*P0.y + P0.z) * TWO_PI) * P0.w;
        float v1 = __sinf((gx*P1.x + gy*P1.y + P1.z) * TWO_PI) * P1.w;
        v2h pk = { (_Float16)v0, (_Float16)v1 };
        *(v2h*)&feat[p*LDS_STRIDE + 2*tid] = pk;
    }
    __syncthreads();

    // ---- GEMM: D[m=pixel, n=kout] = sum_c A[m,c] * W[kout,c] ----
    const int wv    = tid >> 5;
    const int lane  = tid & 31;
    const int r     = lane & 15;
    const int half  = lane >> 4;
    const int kbase = wv * 64;

    v8f acc[2][4];
    #pragma unroll
    for (int mt = 0; mt < 2; ++mt)
        #pragma unroll
        for (int nt = 0; nt < 4; ++nt)
            acc[mt][nt] = (v8f){0.f,0.f,0.f,0.f,0.f,0.f,0.f,0.f};

    #pragma unroll 2
    for (int c0 = 0; c0 < CH; c0 += 32) {
        // A fragment (16x32 f16): lane holds row m=r; VGPR0-3 = K c0+half*16+0..7,
        // VGPR4-7 = K c0+half*16+8..15  -> two b128 LDS loads.
        union { v16h v; v8h h[2]; } afrag[2];
        #pragma unroll
        for (int mt = 0; mt < 2; ++mt) {
            const _Float16* rp = &feat[(mt*16 + r)*LDS_STRIDE + c0 + half*16];
            afrag[mt].h[0] = *(const v8h*)rp;
            afrag[mt].h[1] = *(const v8h*)(rp + 8);
        }
        // B fragment (32x16 f16): lane holds column n=r; VGPR v = K half*16+2v,2v+1
        // -> 16 consecutive f16 from weight row kout starting at c0+half*16.
        v16h bfrag[4];
        #pragma unroll
        for (int nt = 0; nt < 4; ++nt) {
            int kout = kbase + nt*16 + r;
            bfrag[nt] = *(const v16h*)(wh + kout*CH + c0 + half*16);
        }
        #pragma unroll
        for (int mt = 0; mt < 2; ++mt)
            #pragma unroll
            for (int nt = 0; nt < 4; ++nt)
                acc[mt][nt] = __builtin_amdgcn_wmma_f32_16x16x32_f16(
                    false, afrag[mt].v, false, bfrag[nt],
                    (short)0, acc[mt][nt], false, false);
    }

    // ---- store: lane l, VGPR v -> (M = v + 8*half, N = r): 8 consecutive pixels ----
    #pragma unroll
    for (int mt = 0; mt < 2; ++mt) {
        #pragma unroll
        for (int nt = 0; nt < 4; ++nt) {
            int kout  = kbase + nt*16 + r;
            int pbase = mblk*32 + mt*16 + half*8;
            float* op = out + ((size_t)(nb*CH + kout) << 12) + pbase;
            v4f lo = { acc[mt][nt][0], acc[mt][nt][1], acc[mt][nt][2], acc[mt][nt][3] };
            v4f hi = { acc[mt][nt][4], acc[mt][nt][5], acc[mt][nt][6], acc[mt][nt][7] };
            *(v4f*)op       = lo;
            *(v4f*)(op + 4) = hi;
        }
    }
}

extern "C" void kernel_launch(void* const* d_in, const int* in_sizes, int n_in,
                              void* d_out, int out_size, void* d_ws, size_t ws_size,
                              hipStream_t stream) {
    const float* w      = (const float*)d_in[0];
    const float* aw     = (const float*)d_in[1];
    const float* ab     = (const float*)d_in[2];
    const float* weight = (const float*)d_in[3];
    const float* freqs  = (const float*)d_in[4];
    const float* phases = (const float*)d_in[5];

    float4*   params = (float4*)d_ws;                                   // 256 KB
    _Float16* wh     = (_Float16*)((char*)d_ws + (size_t)NBATCH*CH*sizeof(float4)); // 512 KB

    prep_params<<<NBATCH, 256, 0, stream>>>(w, aw, ab, freqs, phases, params);
    conv_weight<<<(CH*CH)/256, 256, 0, stream>>>(weight, wh);
    synth_gemm<<<NBATCH*128, 256, 0, stream>>>(params, wh, (float*)d_out);
}
